// GAT_86251533238780
// MI455X (gfx1250) — compile-verified, matrix operand order checked by reading, not run on hardware
//
#include <hip/hip_runtime.h>
#include <hip/hip_bf16.h>

// ---------------- problem constants (from reference) ----------------
#define NN   20000            // nodes
#define NE   320000           // raw edges
#define ET   (NE + NN)        // edges incl. self loops
#define F0   256              // input feats
#define HD   512              // HEADS*DIM_H (layer1 width)
#define NC   16               // classes (layer2 width)
#define NEG  0.2f

typedef __attribute__((ext_vector_type(16))) _Float16 v16h;
typedef __attribute__((ext_vector_type(8)))  float    v8f;

// ---------------- helpers ----------------
__device__ __forceinline__ float lrelu(float x) { return x > 0.f ? x : NEG * x; }

// order-preserving float <-> u32 map (for atomicMax on floats)
__device__ __forceinline__ unsigned fmap(float x) {
  int i = __float_as_int(x);
  return (i >= 0) ? ((unsigned)i ^ 0x80000000u) : ~(unsigned)i;
}
__device__ __forceinline__ float funmap(unsigned u) {
  int i = (u & 0x80000000u) ? (int)(u ^ 0x80000000u) : (int)~u;
  return __int_as_float(i);
}
#define FMAP_NEG_INF 0x007FFFFFu   // fmap(-inf)

__device__ __forceinline__ void atomAddF(float* p, float v) {
  __hip_atomic_fetch_add(p, v, __ATOMIC_RELAXED, __HIP_MEMORY_SCOPE_AGENT);
}

// edge endpoints incl. implicit self-loops at the tail
__device__ __forceinline__ void edge_sd(const long long* __restrict__ ei, int e, int& s, int& d) {
  if (e < NE) { s = (int)ei[e]; d = (int)ei[NE + e]; }
  else        { s = d = e - NE; }
}

// ---------------- utility kernels ----------------
__global__ void k_fill_u32(unsigned* __restrict__ p, unsigned v, int n) {
  int i = blockIdx.x * blockDim.x + threadIdx.x;
  if (i < n) p[i] = v;
}

__global__ void k_f32_to_h(const float* __restrict__ a, _Float16* __restrict__ b, int n) {
  int i = blockIdx.x * blockDim.x + threadIdx.x;
  if (i < n) b[i] = (_Float16)a[i];
}

// W: K x N (row major, f32)  ->  Wt: N x K (row major, f16)
__global__ void k_transpose_h(const float* __restrict__ W, _Float16* __restrict__ Wt,
                              int K, int N) {
  int i = blockIdx.x * blockDim.x + threadIdx.x;
  if (i >= K * N) return;
  int k = i / N, n = i - k * N;
  Wt[(size_t)n * K + k] = (_Float16)W[i];
}

// ---------------- WMMA GEMM:  C[M,N] = A[M,K] * Bt[N,K]^T + bias ----------------
// One wave per 16x(16*NB) output strip; the A fragment is reused across NB
// v_wmma_f32_16x16x32_f16 issues. Fragment loads are double-buffered so the
// loads for k-step i+1 are in flight while the WMMAs of step i execute.
// Fragment layouts per CDNA5 ISA 7.12.2 (wave32).
template <int NB>
__device__ __forceinline__ void load_frags(const _Float16* __restrict__ arow,
                                           const _Float16* __restrict__ brow,
                                           size_t bstride, int k,
                                           v16h& a, v16h* b) {
  ((float4*)&a)[0] = *(const float4*)(arow + k);        // K halves 0..7 (per-lane)
  ((float4*)&a)[1] = *(const float4*)(arow + k + 16);   // K halves 8..15
  #pragma unroll
  for (int j = 0; j < NB; ++j) {
    ((float4*)&b[j])[0] = *(const float4*)(brow + (size_t)j * bstride + k);
    ((float4*)&b[j])[1] = *(const float4*)(brow + (size_t)j * bstride + k + 16);
  }
}

template <int NB>
__global__ void k_wmma_gemm(const _Float16* __restrict__ A,
                            const _Float16* __restrict__ Bt,
                            const float* __restrict__ bias,
                            float* __restrict__ C,
                            int Mtiles, int NtStrips, int K) {
  const int wavesPerBlock = blockDim.x >> 5;
  int wave = blockIdx.x * wavesPerBlock + (threadIdx.x >> 5);
  if (wave >= Mtiles * NtStrips) return;        // wave-uniform exit (EXEC all-1 for WMMA)
  int mt  = wave / NtStrips;
  int ntb = wave - mt * NtStrips;               // strip of NB consecutive 16-wide N tiles
  int lane  = threadIdx.x & 31;
  int half  = lane >> 4;                         // 0: K 0..7/16..23, 1: K 8..15/24..31
  int idx16 = lane & 15;

  const _Float16* arow = A  + (size_t)(mt * 16 + idx16) * K + half * 8;
  const _Float16* brow = Bt + (size_t)(ntb * NB * 16 + idx16) * K + half * 8;
  const size_t bstride = (size_t)16 * K;         // halves between adjacent N tiles

  v8f acc[NB] = {};
  v16h a0, b0[NB];
  load_frags<NB>(arow, brow, bstride, 0, a0, b0);   // prologue: fragments for k=0

  for (int k = 32; k < K; k += 32) {
    v16h a1, b1[NB];
    load_frags<NB>(arow, brow, bstride, k, a1, b1); // issue next-step loads first
    #pragma unroll
    for (int j = 0; j < NB; ++j)                    // consume previous-step fragments
      acc[j] = __builtin_amdgcn_wmma_f32_16x16x32_f16(
          false, a0, false, b0[j], (short)0, acc[j], false, false);
    a0 = a1;
    #pragma unroll
    for (int j = 0; j < NB; ++j) b0[j] = b1[j];     // register renames, no real moves
  }
  #pragma unroll
  for (int j = 0; j < NB; ++j)                      // epilogue: last k-step
    acc[j] = __builtin_amdgcn_wmma_f32_16x16x32_f16(
        false, a0, false, b0[j], (short)0, acc[j], false, false);

  int N = NtStrips * NB * 16;
  int mBase = mt * 16 + half * 8;                // C/D layout: VGPR r -> M = r (+8 hi lanes)
  #pragma unroll
  for (int j = 0; j < NB; ++j) {
    int n = (ntb * NB + j) * 16 + idx16;
    float bv = bias ? bias[n] : 0.f;
    #pragma unroll
    for (int r = 0; r < 8; ++r)
      C[(size_t)(mBase + r) * N + n] = acc[j][r] + bv;
  }
}

// ---------------- layer-1 edge kernels (H=8, D=64) ----------------
// wave per edge; lane owns 16 contiguous dims -> head = lane>>2
__global__ void k_edge_logits1(const float* __restrict__ xl, const float* __restrict__ xr,
                               const long long* __restrict__ ei,
                               const float* __restrict__ att,  // [8*64] flat
                               float* __restrict__ elog) {
  int e = blockIdx.x * (blockDim.x >> 5) + (threadIdx.x >> 5);
  if (e >= ET) return;
  int lane = threadIdx.x & 31;
  int s, d; edge_sd(ei, e, s, d);
  const float4* pl = (const float4*)(xl + (size_t)s * HD + lane * 16);
  const float4* pr = (const float4*)(xr + (size_t)d * HD + lane * 16);
  const float4* pa = (const float4*)(att + lane * 16);
  float acc = 0.f;
  #pragma unroll
  for (int j = 0; j < 4; ++j) {
    float4 a = pl[j], b = pr[j], t = pa[j];
    acc += lrelu(a.x + b.x) * t.x;
    acc += lrelu(a.y + b.y) * t.y;
    acc += lrelu(a.z + b.z) * t.z;
    acc += lrelu(a.w + b.w) * t.w;
  }
  acc += __shfl_xor(acc, 1);
  acc += __shfl_xor(acc, 2);
  if ((lane & 3) == 0) elog[(size_t)e * 8 + (lane >> 2)] = acc;
}

// generic segment-max / exp-denominator over (edge, head); H = 1<<lgH
__global__ void k_segmax(const float* __restrict__ elog, const long long* __restrict__ ei,
                         unsigned* __restrict__ emax, int lgH) {
  int t = blockIdx.x * blockDim.x + threadIdx.x;
  if (t >= (ET << lgH)) return;
  int e = t >> lgH, h = t & ((1 << lgH) - 1);
  int s, d; edge_sd(ei, e, s, d); (void)s;
  atomicMax(&emax[((size_t)d << lgH) + h], fmap(elog[t]));
}

__global__ void k_expdenom(float* __restrict__ elog, const long long* __restrict__ ei,
                           const unsigned* __restrict__ emax, float* __restrict__ denom,
                           int lgH) {
  int t = blockIdx.x * blockDim.x + threadIdx.x;
  if (t >= (ET << lgH)) return;
  int e = t >> lgH, h = t & ((1 << lgH) - 1);
  int s, d; edge_sd(ei, e, s, d); (void)s;
  float p = __expf(elog[t] - funmap(emax[((size_t)d << lgH) + h]));
  elog[t] = p;
  atomAddF(&denom[((size_t)d << lgH) + h], p);
}

// wave per edge: agg[dst, :] += alpha_h * x_l[src, :]
__global__ void k_aggregate1(const float* __restrict__ p, const float* __restrict__ denom,
                             const float* __restrict__ xl, const long long* __restrict__ ei,
                             float* __restrict__ agg) {
  int e = blockIdx.x * (blockDim.x >> 5) + (threadIdx.x >> 5);
  if (e >= ET) return;
  int lane = threadIdx.x & 31;
  int s, d; edge_sd(ei, e, s, d);
  int h = lane >> 2;
  float alpha = p[(size_t)e * 8 + h] / denom[(size_t)d * 8 + h];
  const float4* ps = (const float4*)(xl + (size_t)s * HD + lane * 16);
  float* po = agg + (size_t)d * HD + lane * 16;
  #pragma unroll
  for (int j = 0; j < 4; ++j) {
    float4 v = ps[j];
    atomAddF(po + 4 * j + 0, alpha * v.x);
    atomAddF(po + 4 * j + 1, alpha * v.y);
    atomAddF(po + 4 * j + 2, alpha * v.z);
    atomAddF(po + 4 * j + 3, alpha * v.w);
  }
}

// h = relu(agg + bias1), stored as f16 for the layer-2 WMMA GEMM
__global__ void k_finalize1(const float* __restrict__ agg, const float* __restrict__ bias1,
                            _Float16* __restrict__ hh, int n) {
  int i = blockIdx.x * blockDim.x + threadIdx.x;
  if (i >= n) return;
  float v = agg[i] + bias1[i & (HD - 1)];
  hh[i] = (_Float16)(v > 0.f ? v : 0.f);
}

// ---------------- layer-2 edge kernels (H=1, D=16) ----------------
__global__ void k_edge_logits2(const float* __restrict__ hl, const float* __restrict__ hr,
                               const long long* __restrict__ ei,
                               const float* __restrict__ att2, float* __restrict__ e2) {
  int e = blockIdx.x * blockDim.x + threadIdx.x;
  if (e >= ET) return;
  int s, d; edge_sd(ei, e, s, d);
  const float4* pl = (const float4*)(hl + (size_t)s * NC);
  const float4* pr = (const float4*)(hr + (size_t)d * NC);
  const float4* pa = (const float4*)att2;
  float acc = 0.f;
  #pragma unroll
  for (int j = 0; j < 4; ++j) {
    float4 a = pl[j], b = pr[j], t = pa[j];
    acc += lrelu(a.x + b.x) * t.x;
    acc += lrelu(a.y + b.y) * t.y;
    acc += lrelu(a.z + b.z) * t.z;
    acc += lrelu(a.w + b.w) * t.w;
  }
  e2[e] = acc;
}

__global__ void k_aggregate2(const float* __restrict__ p, const float* __restrict__ denom,
                             const float* __restrict__ hl, const long long* __restrict__ ei,
                             float* __restrict__ out) {
  int e = blockIdx.x * blockDim.x + threadIdx.x;
  if (e >= ET) return;
  int s, d; edge_sd(ei, e, s, d);
  float alpha = p[e] / denom[d];
  const float* ps = hl + (size_t)s * NC;
  float* po = out + (size_t)d * NC;
  #pragma unroll
  for (int c = 0; c < NC; ++c) atomAddF(po + c, alpha * ps[c]);
}

__global__ void k_bias_out(float* __restrict__ out, const float* __restrict__ bias2, int n) {
  int i = blockIdx.x * blockDim.x + threadIdx.x;
  if (i < n) out[i] += bias2[i & (NC - 1)];
}

// ---------------- host launcher ----------------
extern "C" void kernel_launch(void* const* d_in, const int* in_sizes, int n_in,
                              void* d_out, int out_size, void* d_ws, size_t ws_size,
                              hipStream_t stream) {
  const float*     x     = (const float*)d_in[0];
  const long long* ei    = (const long long*)d_in[1];   // int64 [2, NE]
  const float*     W_l1  = (const float*)d_in[2];
  const float*     b_l1  = (const float*)d_in[3];
  const float*     W_r1  = (const float*)d_in[4];
  const float*     b_r1  = (const float*)d_in[5];
  const float*     att1  = (const float*)d_in[6];
  const float*     bias1 = (const float*)d_in[7];
  const float*     W_l2  = (const float*)d_in[8];
  const float*     b_l2  = (const float*)d_in[9];
  const float*     W_r2  = (const float*)d_in[10];
  const float*     b_r2  = (const float*)d_in[11];
  const float*     att2  = (const float*)d_in[12];
  const float*     bias2 = (const float*)d_in[13];
  float* out = (float*)d_out;

  // linear workspace carve-out
  char* w = (char*)d_ws;
  auto alloc = [&](size_t bytes) -> char* {
    char* p = w; w += (bytes + 255) & ~(size_t)255; return p;
  };
  _Float16* xh     = (_Float16*)alloc((size_t)NN * F0 * 2);   // x as f16
  _Float16* wt1l   = (_Float16*)alloc((size_t)HD * F0 * 2);   // W_l1^T f16 [512,256]
  _Float16* wt1r   = (_Float16*)alloc((size_t)HD * F0 * 2);
  float*    xl1    = (float*)alloc((size_t)NN * HD * 4);
  float*    xr1    = (float*)alloc((size_t)NN * HD * 4);
  float*    elog1  = (float*)alloc((size_t)ET * 8 * 4);
  unsigned* emax1  = (unsigned*)alloc((size_t)NN * 8 * 4);
  float*    den1   = (float*)alloc((size_t)NN * 8 * 4);
  float*    agg1   = (float*)alloc((size_t)NN * HD * 4);
  _Float16* hh     = (_Float16*)alloc((size_t)NN * HD * 2);   // relu(h) f16
  _Float16* wt2l   = (_Float16*)alloc((size_t)NC * HD * 2);   // W_l2^T f16 [16,512]
  _Float16* wt2r   = (_Float16*)alloc((size_t)NC * HD * 2);
  float*    hl2    = (float*)alloc((size_t)NN * NC * 4);
  float*    hr2    = (float*)alloc((size_t)NN * NC * 4);
  float*    e2     = (float*)alloc((size_t)ET * 4);
  unsigned* emax2  = (unsigned*)alloc((size_t)NN * 4);
  float*    den2   = (float*)alloc((size_t)NN * 4);
  (void)ws_size; (void)in_sizes; (void)n_in; (void)out_size;

  auto cdiv = [](int a, int b) { return (a + b - 1) / b; };
  const int T = 256;  // 8 wave32 per block

  // --- precision conversion + weight transposes ---
  k_f32_to_h   <<<cdiv(NN * F0, T), T, 0, stream>>>(x, xh, NN * F0);
  k_transpose_h<<<cdiv(F0 * HD, T), T, 0, stream>>>(W_l1, wt1l, F0, HD);
  k_transpose_h<<<cdiv(F0 * HD, T), T, 0, stream>>>(W_r1, wt1r, F0, HD);
  k_transpose_h<<<cdiv(HD * NC, T), T, 0, stream>>>(W_l2, wt2l, HD, NC);
  k_transpose_h<<<cdiv(HD * NC, T), T, 0, stream>>>(W_r2, wt2r, HD, NC);

  // --- layer 1 GEMMs: x_l1 = x@W_l1+b, x_r1 = x@W_r1+b (WMMA f16->f32) ---
  {
    const int NB = 4;                             // 16x64 strip per wave
    int Mt = NN / 16, NtS = (HD / 16) / NB;       // 1250 x 8 strips
    int waves = Mt * NtS, blocks = cdiv(waves, T / 32);
    k_wmma_gemm<NB><<<blocks, T, 0, stream>>>(xh, wt1l, b_l1, xl1, Mt, NtS, F0);
    k_wmma_gemm<NB><<<blocks, T, 0, stream>>>(xh, wt1r, b_r1, xr1, Mt, NtS, F0);
  }

  // --- init accumulators ---
  k_fill_u32<<<cdiv(NN * 8, T), T, 0, stream>>>(emax1, FMAP_NEG_INF, NN * 8);
  k_fill_u32<<<cdiv(NN * 8, T), T, 0, stream>>>((unsigned*)den1, 0u, NN * 8);
  k_fill_u32<<<cdiv(NN * HD, T), T, 0, stream>>>((unsigned*)agg1, 0u, NN * HD);

  // --- layer 1 edge phase ---
  int ewaveBlocks = cdiv(ET, T / 32);
  k_edge_logits1<<<ewaveBlocks, T, 0, stream>>>(xl1, xr1, ei, att1, elog1);
  k_segmax     <<<cdiv(ET * 8, T), T, 0, stream>>>(elog1, ei, emax1, 3);
  k_expdenom   <<<cdiv(ET * 8, T), T, 0, stream>>>(elog1, ei, emax1, den1, 3);
  k_aggregate1 <<<ewaveBlocks, T, 0, stream>>>(elog1, den1, xl1, ei, agg1);
  k_finalize1  <<<cdiv(NN * HD, T), T, 0, stream>>>(agg1, bias1, hh, NN * HD);

  // --- layer 2 GEMMs: hl2 = h@W_l2+b, hr2 = h@W_r2+b ---
  {
    int Mt = NN / 16, NtS = NC / 16;              // 1250 x 1 tiles
    int waves = Mt * NtS, blocks = cdiv(waves, T / 32);
    k_wmma_gemm<1><<<blocks, T, 0, stream>>>(hh, wt2l, b_l2, hl2, Mt, NtS, HD);
    k_wmma_gemm<1><<<blocks, T, 0, stream>>>(hh, wt2r, b_r2, hr2, Mt, NtS, HD);
  }

  // --- layer 2 edge phase (accumulate straight into d_out) ---
  k_fill_u32<<<cdiv(NN, T), T, 0, stream>>>(emax2, FMAP_NEG_INF, NN);
  k_fill_u32<<<cdiv(NN, T), T, 0, stream>>>((unsigned*)den2, 0u, NN);
  k_fill_u32<<<cdiv(NN * NC, T), T, 0, stream>>>((unsigned*)out, 0u, NN * NC);

  k_edge_logits2<<<cdiv(ET, T), T, 0, stream>>>(hl2, hr2, ei, att2, e2);
  k_segmax      <<<cdiv(ET, T), T, 0, stream>>>(e2, ei, emax2, 0);
  k_expdenom    <<<cdiv(ET, T), T, 0, stream>>>(e2, ei, emax2, den2, 0);
  k_aggregate2  <<<cdiv(ET, T), T, 0, stream>>>(e2, den2, hl2, ei, out);
  k_bias_out    <<<cdiv(NN * NC, T), T, 0, stream>>>(out, bias2, NN * NC);
}
